// HCFN_9818295238778
// MI455X (gfx1250) — matrix operand order repaired and checked by reading, not compiled, not run on hardware
//
#include <hip/hip_runtime.h>
#include <hip/hip_bf16.h>
#include <cstdint>
#include <cstddef>

// ---------------------------------------------------------------------------
// AFTCNBlock forward on MI455X (gfx1250), bf16 WMMA for all GEMMs.
// Shapes: B=32, L=1024, C_IN=64, F=256, K=8, R=4, H=32, DEPTH=3.
// ---------------------------------------------------------------------------

typedef __bf16 bf16;
typedef __attribute__((ext_vector_type(16))) __bf16 v16bf;
typedef __attribute__((ext_vector_type(8)))  __bf16 v8bf;
typedef __attribute__((ext_vector_type(4)))  __bf16 v4bf;
typedef __attribute__((ext_vector_type(8)))  float  v8f;

#define WMMA_BF16(a, b, c) \
  __builtin_amdgcn_wmma_f32_16x16x32_bf16(false, (a), false, (b), (short)0, (c), false, false)

static constexpr int Bn = 32;
static constexpr int Ln = 1024;
static constexpr int Fn = 256;
static constexpr int Kt = 8;     // taps
static constexpr int Hn = 32;    // controller hidden
static constexpr int BL = Bn * Ln;
static constexpr float EPSf = 1e-5f;

// ---------------------------------------------------------------------------
// Weight packing into WMMA B-fragment order.
// B fragment for tile (t = k-tile of 32, ct = 16-column tile):
//   lane 0..31, 16 bf16 elems j: k = t*32 + (lane>>4)*16 + j, n = ct*16 + (lane&15)
// Stored contiguously: frag[((t*CT + ct)*32 + lane)*16 + j]  (32B per lane)
// ---------------------------------------------------------------------------

// conv kernels f32 [R,256,Cin,8] -> frag with col = r*256+f, k = c*8+kk, CT=64
__global__ void pack_conv_kernel(const float* __restrict__ kern,
                                 bf16* __restrict__ out, int Cin, int total) {
  int idx = blockIdx.x * blockDim.x + threadIdx.x;
  if (idx >= total) return;
  int j    = idx & 15;
  int lane = (idx >> 4) & 31;
  int ct   = (idx >> 9) & 63;
  int t    = idx >> 15;
  int k    = t * 32 + ((lane >> 4) << 4) + j;
  int col  = ct * 16 + (lane & 15);
  int r = col >> 8, f = col & 255;
  int c = k >> 3, kk = k & 7;
  out[idx] = (bf16)kern[(((size_t)r * Fn + f) * Cin + c) * Kt + kk];
}

// generic source: val = src[k*sk + col*sc]
__global__ void pack_mat_kernel(const float* __restrict__ src,
                                bf16* __restrict__ out,
                                int CT, int sk, int sc, int total) {
  int idx = blockIdx.x * blockDim.x + threadIdx.x;
  if (idx >= total) return;
  int j    = idx & 15;
  int lane = (idx >> 4) & 31;
  int rest = idx >> 9;
  int ct   = rest % CT;
  int t    = rest / CT;
  int k    = t * 32 + ((lane >> 4) << 4) + j;
  int col  = ct * 16 + (lane & 15);
  out[idx] = (bf16)src[(size_t)k * sk + (size_t)col * sc];
}

__global__ void convert_bf16_kernel(const float* __restrict__ in,
                                    bf16* __restrict__ out, int n) {
  int i = blockIdx.x * blockDim.x + threadIdx.x;
  if (i < n) out[i] = (bf16)in[i];
}

__global__ void zero_kernel(float* __restrict__ p, int n) {
  int i = blockIdx.x * blockDim.x + threadIdx.x;
  if (i < n) p[i] = 0.f;
}

// ---------------------------------------------------------------------------
// Patch GEMM (controller layer-1 and the 1x1 downsample).
// A[row=b*L+l, c*TAPS+kk] = act[b, l-(TAPS-1-kk), c]  (dilation 1, zero pad)
// One wave per 16l x 16n tile. TAPS = 8 (controller) or 1 (downsample).
// ---------------------------------------------------------------------------
template <int TAPS>
__global__ __launch_bounds__(32)
void gemm_patch_wmma(const bf16* __restrict__ act,
                     const bf16* __restrict__ Bfrag,
                     const float* __restrict__ bias,   // [N] or nullptr
                     float* __restrict__ out,          // [B*L, N]
                     int Cin, int N, int relu) {
  const int l0   = blockIdx.x * 16;
  const int nt   = blockIdx.y;
  const int b    = blockIdx.z;
  const int lane = threadIdx.x;
  const int CT   = N >> 4;
  const int T    = (Cin * TAPS) >> 5;

  const int m  = lane & 15;
  const int hi = lane >> 4;
  const int l  = l0 + m;
  const bf16* actb = act + (size_t)b * Ln * Cin;

  v8f acc = {};
  for (int t = 0; t < T; ++t) {
    const int kb = t * 32 + hi * 16;
    v16bf a;
#pragma unroll
    for (int j = 0; j < 16; ++j) {
      int k = kb + j;
      int c, lsrc;
      if (TAPS == 1) { c = k; lsrc = l; }
      else { c = k >> 3; int kk = k & 7; lsrc = l - (TAPS - 1 - kk); }
      a[j] = (lsrc >= 0) ? actb[(size_t)lsrc * Cin + c] : (bf16)0.0f;
    }
    v16bf bv = *(const v16bf*)(Bfrag + (((size_t)t * CT + nt) * 32 + lane) * 16);
    acc = WMMA_BF16(a, bv, acc);
  }

  const int n = nt * 16 + m;
  const float bv = bias ? bias[n] : 0.f;
#pragma unroll
  for (int v = 0; v < 8; ++v) {
    int lr = l0 + v + hi * 8;
    float val = acc[v] + bv;
    if (relu) val = fmaxf(val, 0.f);
    out[((size_t)b * Ln + lr) * N + n] = val;
  }
}

// ---------------------------------------------------------------------------
// Fuzzy causal conv as WMMA GEMM with fused rule-softmax reduction.
// 4 waves/WG, each wave: 16l x 16f tile with 4 accumulators (r=0..3).
// The dilated causal patch matrix is staged in LDS in im2col (k-ordered)
// layout so each A fragment is two contiguous ds_load_b128 per iteration.
// ---------------------------------------------------------------------------
__global__ __launch_bounds__(128)
void fuzzy_conv_wmma(const bf16* __restrict__ act,    // [B, L, Cin]
                     const bf16* __restrict__ Wfrag,  // packed, CT=64
                     const float* __restrict__ bias,  // [R*256]
                     const float* __restrict__ frs,   // [B*L, 4]
                     float* __restrict__ out,         // [B*L, 256]
                     int Cin, int dil) {
  extern __shared__ char smem_raw[];
  bf16* tileA = (bf16*)smem_raw;           // [16][Cin*8] im2col, k-ordered

  const int l0   = blockIdx.x * 16;
  const int fgrp = blockIdx.y;             // 0..3
  const int b    = blockIdx.z;
  const int tid  = threadIdx.x;
  const int wave = tid >> 5;
  const int lane = tid & 31;
  const int Kr   = Cin * Kt;

  // ---- im2col fill: 128 (m,kk) row-pairs, each Cin contiguous elements ----
  // tileA[m*Kr + c*8 + kk] = act[b, l0+m-(7-kk)*dil, c]  (zero if OOB left)
  {
    const bf16* actb = act + (size_t)b * Ln * Cin;
    const int cq4   = Cin >> 2;             // 4-element chunks per row
    const int shift = 31 - __clz(cq4);      // cq4 is a power of two (16 or 64)
    const int total = 128 << shift;
    for (int ci = tid; ci < total; ci += 128) {
      int pair = ci >> shift;               // 0..127
      int cq   = ci & (cq4 - 1);
      int m = pair >> 3, kk = pair & 7;
      int g  = l0 + m - (7 - kk) * dil;
      int c0 = cq << 2;
      v4bf v;
      if (g >= 0) v = *(const v4bf*)(actb + (size_t)g * Cin + c0);
      else        v = (v4bf)((__bf16)0.0f);
      bf16* dst = tileA + (size_t)m * Kr + c0 * 8 + kk;
      dst[0] = v[0]; dst[8] = v[1]; dst[16] = v[2]; dst[24] = v[3];
    }
  }
  __syncthreads();

  const int ft = fgrp * 4 + wave;          // 0..15 (16-wide f tile)
  const int T  = Kr >> 5;
  const int m  = lane & 15;
  const int hi = lane >> 4;
  const bf16* arow = tileA + (size_t)m * Kr;

  v8f acc0 = {}, acc1 = {}, acc2 = {}, acc3 = {};
  for (int t = 0; t < T; ++t) {
    const int kb = t * 32 + hi * 16;
    v8bf alo = *(const v8bf*)(arow + kb);        // k = kb..kb+7
    v8bf ahi = *(const v8bf*)(arow + kb + 8);    // k = kb+8..kb+15
    v16bf a;
#pragma unroll
    for (int j = 0; j < 8; ++j) { a[j] = alo[j]; a[j + 8] = ahi[j]; }

    const bf16* wp = Wfrag + (size_t)t * (64 * 32 * 16) + (size_t)lane * 16;
    __builtin_prefetch(wp + (size_t)(64 * 32 * 16) + (size_t)ft * 512, 0, 0);
    v16bf b0 = *(const v16bf*)(wp + (size_t)(0 * 16 + ft) * 512);
    v16bf b1 = *(const v16bf*)(wp + (size_t)(1 * 16 + ft) * 512);
    v16bf b2 = *(const v16bf*)(wp + (size_t)(2 * 16 + ft) * 512);
    v16bf b3 = *(const v16bf*)(wp + (size_t)(3 * 16 + ft) * 512);
    acc0 = WMMA_BF16(a, b0, acc0);
    acc1 = WMMA_BF16(a, b1, acc1);
    acc2 = WMMA_BF16(a, b2, acc2);
    acc3 = WMMA_BF16(a, b3, acc3);
  }

  const int f = ft * 16 + m;
  const float bb0 = bias[0 * 256 + f];
  const float bb1 = bias[1 * 256 + f];
  const float bb2 = bias[2 * 256 + f];
  const float bb3 = bias[3 * 256 + f];
#pragma unroll
  for (int v = 0; v < 8; ++v) {
    int lr = l0 + v + hi * 8;
    const float4 fr = *(const float4*)(frs + ((size_t)b * Ln + lr) * 4);
    float val = fr.x * (acc0[v] + bb0) + fr.y * (acc1[v] + bb1) +
                fr.z * (acc2[v] + bb2) + fr.w * (acc3[v] + bb3);
    out[((size_t)b * Ln + lr) * 256 + f] = val;
  }
}

// ---------------------------------------------------------------------------
// Controller stage 2: frs = softmax(hid @ cw2 + cb2), hid is post-ReLU [BL,32]
// ---------------------------------------------------------------------------
__global__ void ctrl_frs_kernel(const float* __restrict__ hid,
                                const float* __restrict__ cw2,  // [32,4]
                                const float* __restrict__ cb2,  // [4]
                                float* __restrict__ frs) {
  int row = blockIdx.x * blockDim.x + threadIdx.x;
  if (row >= BL) return;
  const float* h = hid + (size_t)row * Hn;
  float lg0 = cb2[0], lg1 = cb2[1], lg2 = cb2[2], lg3 = cb2[3];
#pragma unroll
  for (int j = 0; j < Hn; ++j) {
    float s = h[j];
    lg0 += s * cw2[j * 4 + 0];
    lg1 += s * cw2[j * 4 + 1];
    lg2 += s * cw2[j * 4 + 2];
    lg3 += s * cw2[j * 4 + 3];
  }
  float mx = fmaxf(fmaxf(lg0, lg1), fmaxf(lg2, lg3));
  float e0 = __expf(lg0 - mx), e1 = __expf(lg1 - mx);
  float e2 = __expf(lg2 - mx), e3 = __expf(lg3 - mx);
  float inv = 1.f / (e0 + e1 + e2 + e3);
  float* o = frs + (size_t)row * 4;
  o[0] = e0 * inv; o[1] = e1 * inv; o[2] = e2 * inv; o[3] = e3 * inv;
}

// ---------------------------------------------------------------------------
// BatchNorm batch statistics (training mode): per-channel sum / sumsq.
// ---------------------------------------------------------------------------
__global__ __launch_bounds__(256)
void bn_stats_kernel(const float* __restrict__ h,  // [BL, 256]
                     float* __restrict__ stats) {  // [512]: sum, sumsq
  int ch = threadIdx.x;
  int r0 = blockIdx.x * 256;
  float s = 0.f, ss = 0.f;
  for (int r = 0; r < 256; ++r) {
    float v = h[((size_t)(r0 + r)) * 256 + ch];
    s += v; ss += v * v;
  }
  atomicAdd(&stats[ch], s);
  atomicAdd(&stats[256 + ch], ss);
}

__device__ __forceinline__ float elu1(float x) { return x > 0.f ? x : expm1f(x); }

// mid-layer: act_out = bf16(elu(bn(h)))
__global__ void bn_elu_act_kernel(const float* __restrict__ h,
                                  const float* __restrict__ stats,
                                  const float* __restrict__ g,
                                  const float* __restrict__ be,
                                  bf16* __restrict__ act_out) {
  int idx = blockIdx.x * blockDim.x + threadIdx.x;
  if (idx >= BL * 256) return;
  int ch = idx & 255;
  const float invN = 1.f / (float)BL;
  float mean = stats[ch] * invN;
  float var  = stats[256 + ch] * invN - mean * mean;
  float y = g[ch] * (h[idx] - mean) * rsqrtf(var + EPSf) + be[ch];
  act_out[idx] = (bf16)elu1(y);
}

// block tail: out = elu(elu(bn(h)) + res); write res(f32) + act(bf16)
__global__ void bn_elu_res_kernel(const float* __restrict__ h,
                                  const float* __restrict__ stats,
                                  const float* __restrict__ g,
                                  const float* __restrict__ be,
                                  const float* __restrict__ res_in,
                                  float* __restrict__ res_out,
                                  bf16* __restrict__ act_out) {
  int idx = blockIdx.x * blockDim.x + threadIdx.x;
  if (idx >= BL * 256) return;
  int ch = idx & 255;
  const float invN = 1.f / (float)BL;
  float mean = stats[ch] * invN;
  float var  = stats[256 + ch] * invN - mean * mean;
  float y = g[ch] * (h[idx] - mean) * rsqrtf(var + EPSf) + be[ch];
  y = elu1(y);
  float o = elu1(y + res_in[idx]);
  res_out[idx] = o;
  act_out[idx] = (bf16)o;
}

// ---------------------------------------------------------------------------
// Host-side orchestration
// ---------------------------------------------------------------------------
extern "C" void kernel_launch(void* const* d_in, const int* in_sizes, int n_in,
                              void* d_out, int out_size, void* d_ws, size_t ws_size,
                              hipStream_t stream) {
  if (n_in < 34) return;

  const float *x, *dw;
  const float *ci_w1, *ci_b1, *ci_w2, *ci_b2;
  const float *cf_w1, *cf_b1, *cf_w2, *cf_b2;
  const float *k1[3], *b1[3], *g1[3], *be1[3], *k2[3], *b2[3], *g2[3], *be2[3];
  auto P = [&](int i) { return (const float*)d_in[i]; };

  if (in_sizes[0] == 65536) {
    // jax tree-flatten order (dict keys sorted)
    cf_w1 = P(0); cf_b1 = P(1); cf_w2 = P(2); cf_b2 = P(3);
    ci_w1 = P(4); ci_b1 = P(5); ci_w2 = P(6); ci_b2 = P(7);
    dw = P(8);
    for (int d = 0; d < 3; ++d) {
      int s = 9 + d * 8;   // b1,b2,be1,be2,g1,g2,k1,k2
      b1[d] = P(s + 0); b2[d] = P(s + 1); be1[d] = P(s + 2); be2[d] = P(s + 3);
      g1[d] = P(s + 4); g2[d] = P(s + 5); k1[d] = P(s + 6);  k2[d] = P(s + 7);
    }
    x = P(33);
  } else {
    // dict insertion order
    x = P(0); dw = P(1);
    ci_w1 = P(2); ci_b1 = P(3); ci_w2 = P(4); ci_b2 = P(5);
    cf_w1 = P(6); cf_b1 = P(7); cf_w2 = P(8); cf_b2 = P(9);
    for (int d = 0; d < 3; ++d) {
      int s = 10 + d * 8;  // k1,b1,g1,be1,k2,b2,g2,be2
      k1[d] = P(s + 0); b1[d] = P(s + 1); g1[d] = P(s + 2); be1[d] = P(s + 3);
      k2[d] = P(s + 4); b2[d] = P(s + 5); g2[d] = P(s + 6); be2[d] = P(s + 7);
    }
  }

  // ---- workspace bump allocator -------------------------------------------
  char* ws = (char*)d_ws;
  size_t off = 0;
  auto alloc = [&](size_t bytes) -> void* {
    void* p = ws + off;
    off = (off + bytes + 255) & ~(size_t)255;
    return p;
  };
  bf16*  act0   = (bf16*)alloc((size_t)BL * 64 * 2);
  bf16*  actA   = (bf16*)alloc((size_t)BL * 256 * 2);
  bf16*  actB   = (bf16*)alloc((size_t)BL * 256 * 2);
  float* res    = (float*)alloc((size_t)BL * 256 * 4);
  float* hbuf   = (float*)alloc((size_t)BL * 256 * 4);
  float* hid    = (float*)alloc((size_t)BL * Hn * 4);
  float* frs    = (float*)alloc((size_t)BL * 4 * 4);
  float* stats  = (float*)alloc(512 * 4);
  bf16*  dwfrag   = (bf16*)alloc((size_t)2 * 16 * 32 * 16 * 2);
  bf16*  cw1ifrag = (bf16*)alloc((size_t)16 * 2 * 32 * 16 * 2);
  bf16*  cw1ffrag = (bf16*)alloc((size_t)64 * 2 * 32 * 16 * 2);
  bf16*  kfrag[3][2];
  for (int d = 0; d < 3; ++d) {
    int T1 = (d == 0) ? 16 : 64;
    kfrag[d][0] = (bf16*)alloc((size_t)T1 * 64 * 32 * 16 * 2);
    kfrag[d][1] = (bf16*)alloc((size_t)64 * 64 * 32 * 16 * 2);
  }
  if (off > ws_size) return;

  // ---- pack weights into WMMA fragment order ------------------------------
  {
    int tot = 2 * 16 * 32 * 16;   // downsample_w [256,64]: k stride 1, col stride 64
    pack_mat_kernel<<<tot / 256, 256, 0, stream>>>(dw, dwfrag, 16, 1, 64, tot);
    tot = 16 * 2 * 32 * 16;       // ctrl_in cw1 [512,32]
    pack_mat_kernel<<<tot / 256, 256, 0, stream>>>(ci_w1, cw1ifrag, 2, 32, 1, tot);
    tot = 64 * 2 * 32 * 16;       // ctrl_f cw1 [2048,32]
    pack_mat_kernel<<<tot / 256, 256, 0, stream>>>(cf_w1, cw1ffrag, 2, 32, 1, tot);
    for (int d = 0; d < 3; ++d) {
      int Cin1 = (d == 0) ? 64 : 256;
      int t1 = ((Cin1 * 8) >> 5) * 64 * 32 * 16;
      pack_conv_kernel<<<t1 / 256, 256, 0, stream>>>(k1[d], kfrag[d][0], Cin1, t1);
      int t2 = 64 * 64 * 32 * 16;
      pack_conv_kernel<<<t2 / 256, 256, 0, stream>>>(k2[d], kfrag[d][1], 256, t2);
    }
  }

  // ---- x -> bf16 [B,L,64]; downsample residual (WMMA GEMM) ----------------
  convert_bf16_kernel<<<(BL * 64) / 256, 256, 0, stream>>>(x, act0, BL * 64);
  gemm_patch_wmma<1><<<dim3(Ln / 16, 16, Bn), 32, 0, stream>>>(
      act0, dwfrag, nullptr, res, 64, 256, 0);

  // ---- layers -------------------------------------------------------------
  const int nblk = (BL * 256) / 256;
  for (int d = 0; d < 3; ++d) {
    int dil  = 1 << d;
    int Cin1 = (d == 0) ? 64 : 256;
    const bf16* act_in = (d == 0) ? act0 : actB;
    const bf16* cw1p   = (d == 0) ? cw1ifrag : cw1ffrag;
    const float* cb1p  = (d == 0) ? ci_b1 : cf_b1;
    const float* cw2p  = (d == 0) ? ci_w2 : cf_w2;
    const float* cb2p  = (d == 0) ? ci_b2 : cf_b2;

    // controller 1 (WMMA) + softmax head
    gemm_patch_wmma<8><<<dim3(Ln / 16, 2, Bn), 32, 0, stream>>>(
        act_in, cw1p, cb1p, hid, Cin1, Hn, 1);
    ctrl_frs_kernel<<<BL / 256, 256, 0, stream>>>(hid, cw2p, cb2p, frs);

    // fuzzy conv 1 (WMMA, fused rule mix) -> hbuf
    {
      size_t sh = (size_t)16 * Cin1 * Kt * sizeof(bf16);
      fuzzy_conv_wmma<<<dim3(Ln / 16, 4, Bn), 128, sh, stream>>>(
          act_in, kfrag[d][0], b1[d], frs, hbuf, Cin1, dil);
    }
    zero_kernel<<<2, 256, 0, stream>>>(stats, 512);
    bn_stats_kernel<<<BL / 256, 256, 0, stream>>>(hbuf, stats);
    bn_elu_act_kernel<<<nblk, 256, 0, stream>>>(hbuf, stats, g1[d], be1[d], actA);

    // controller (always ctrl_f) for conv 2
    gemm_patch_wmma<8><<<dim3(Ln / 16, 2, Bn), 32, 0, stream>>>(
        actA, cw1ffrag, cf_b1, hid, 256, Hn, 1);
    ctrl_frs_kernel<<<BL / 256, 256, 0, stream>>>(hid, cf_w2, cf_b2, frs);

    // fuzzy conv 2 -> hbuf
    {
      size_t sh = (size_t)16 * 256 * Kt * sizeof(bf16);
      fuzzy_conv_wmma<<<dim3(Ln / 16, 4, Bn), 128, sh, stream>>>(
          actA, kfrag[d][1], b2[d], frs, hbuf, 256, dil);
    }
    zero_kernel<<<2, 256, 0, stream>>>(stats, 512);
    bn_stats_kernel<<<BL / 256, 256, 0, stream>>>(hbuf, stats);

    float* res_out = (d == 2) ? (float*)d_out : res;
    bn_elu_res_kernel<<<nblk, 256, 0, stream>>>(hbuf, stats, g2[d], be2[d],
                                                res, res_out, actB);
  }
  (void)out_size; (void)in_sizes;
}
